// DilatedMHCABlock_25512105738921
// MI455X (gfx1250) — compile-verified, matrix operand order checked by reading, not run on hardware
//
#include <hip/hip_runtime.h>
#include <math.h>

// ---------------- problem constants ----------------
#define Bq   4
#define Nseq 2048
#define Dm   1024
#define Hh   16
#define DH   64           // Dm / Hh
#define Mtot (Bq * Nseq)  // 8192
#define KW   (Dm / 2)     // 512 packed bf16-pair words per row

// GEMM tiling
#define TM   256          // workgroup tile M
#define TN   128          // workgroup tile N
#define ROWW 20           // LDS row stride in words (16 data + 4 pad; 16 lanes * 20 -> 16 banks)
#define A_WORDS (TM * ROWW)   // 5120 words per buffer
#define B_WORDS (TN * ROWW)   // 2560 words per buffer

typedef __attribute__((ext_vector_type(16))) __bf16 v16bf;
typedef __attribute__((ext_vector_type(8)))  float  v8f;

union FragU {
    uint4 q[2];
    v16bf v;
};

__device__ __forceinline__ unsigned int pack_bf16_rne(float a, float b) {
    unsigned int ua = __float_as_uint(a);
    unsigned int ub = __float_as_uint(b);
    ua += 0x7FFFu + ((ua >> 16) & 1u);
    ub += 0x7FFFu + ((ub >> 16) & 1u);
    return (ua >> 16) | (ub & 0xFFFF0000u);
}

// ---------------------------------------------------------------------------
// Pack x (fp32, M x 1024 row-major) -> bf16 pair words xp[m*KW + c]
// ---------------------------------------------------------------------------
__global__ void dmhca_pack_x(const float* __restrict__ x, unsigned int* __restrict__ xp) {
    int idx = blockIdx.x * blockDim.x + threadIdx.x;       // over Mtot*KW
    int m = idx >> 9;
    int c = idx & (KW - 1);
    size_t g = (size_t)m * Dm + 2 * c;
    xp[idx] = pack_bf16_rne(x[g], x[g + 1]);
}

// ---------------------------------------------------------------------------
// Pack weight W (fp32, K x N row-major) column-major into bf16 pairs:
//   wp[n*KW + c] = (W[2c][n], W[2c+1][n]).  grid.z selects which of 4 weights.
// ---------------------------------------------------------------------------
__global__ void dmhca_pack_w(const float* __restrict__ w0, const float* __restrict__ w1,
                             const float* __restrict__ w2, const float* __restrict__ w3,
                             unsigned int* __restrict__ wp) {
    int z = blockIdx.z;
    const float* W = (z == 0) ? w0 : (z == 1) ? w1 : (z == 2) ? w2 : w3;
    unsigned int* Op = wp + (size_t)z * Dm * KW;
    int idx = blockIdx.x * blockDim.x + threadIdx.x;       // over Dm*KW
    int c = idx >> 10;
    int n = idx & (Dm - 1);
    float a = W[(size_t)(2 * c) * Dm + n];
    float b = W[(size_t)(2 * c + 1) * Dm + n];
    Op[(size_t)n * KW + c] = pack_bf16_rne(a, b);
}

// ---------------------------------------------------------------------------
// Tiled bf16 WMMA GEMM with async global->LDS tile copies (double buffered):
//   out[M x 1024] = A[M x 1024] * B[1024 x 1024] + bias
// Workgroup 256 thr = 8 wave32; tile 256x128; wave tile 64x64 (16 WMMAs/K-step).
// ---------------------------------------------------------------------------
__global__ void __launch_bounds__(256)
dmhca_gemm_bf16(const unsigned int* __restrict__ Ap, const unsigned int* __restrict__ Bp,
                const float* __restrict__ bias, float* __restrict__ out) {
    __shared__ unsigned int ldsA[2][A_WORDS];
    __shared__ unsigned int ldsB[2][B_WORDS];

    const int tid  = threadIdx.x;
    const int lane = tid & 31;
    const int wave = tid >> 5;           // 0..7
    const int wm   = wave & 3;           // 4 waves along M (64 each)
    const int wn   = wave >> 2;          // 2 waves along N (64 each)
    const int l15  = lane & 15;
    const int half = lane >> 4;

    const int N0 = blockIdx.x * TN;
    const int M0 = blockIdx.y * TM;

    // --- per-thread async-copy slots -------------------------------------
    // A tile: 256 rows x 4 x 16B chunks = 1024 chunks -> 4 per thread
    // B tile: 128 rows x 4 chunks       =  512 chunks -> 2 per thread
    const unsigned int* gA[4];  unsigned lA[4];
    const unsigned int* gB[2];  unsigned lB[2];
#pragma unroll
    for (int i = 0; i < 4; ++i) {
        int idx = tid + i * 256;
        int row = idx >> 2, q = idx & 3;
        gA[i] = Ap + (size_t)(M0 + row) * KW + q * 4;
        lA[i] = (unsigned)(uintptr_t)&ldsA[0][row * ROWW + q * 4];
    }
#pragma unroll
    for (int i = 0; i < 2; ++i) {
        int idx = tid + i * 256;
        int row = idx >> 2, q = idx & 3;
        gB[i] = Bp + (size_t)(N0 + row) * KW + q * 4;
        lB[i] = (unsigned)(uintptr_t)&ldsB[0][row * ROWW + q * 4];
    }

    v8f acc[16];
#pragma unroll
    for (int i = 0; i < 16; ++i)
#pragma unroll
        for (int r = 0; r < 8; ++r) acc[i][r] = 0.0f;

    // prologue: async-copy K-step 0 into buffer 0
#pragma unroll
    for (int i = 0; i < 4; ++i)
        asm volatile("global_load_async_to_lds_b128 %0, %1, off"
                     :: "v"(lA[i]), "v"(gA[i]) : "memory");
#pragma unroll
    for (int i = 0; i < 2; ++i)
        asm volatile("global_load_async_to_lds_b128 %0, %1, off"
                     :: "v"(lB[i]), "v"(gB[i]) : "memory");

    for (int kk = 0; kk < 32; ++kk) {           // 32 K-steps of 32 => K = 1024
        const int cur = kk & 1;
        // my async copies done -> barrier -> everyone's copies done,
        // and everyone finished reading the buffer we are about to refill.
        asm volatile("s_wait_asynccnt 0x0" ::: "memory");
        __syncthreads();

        if (kk + 1 < 32) {                      // async-fill the other buffer
            const unsigned aOff = (1 - cur) * (A_WORDS * 4);
            const unsigned bOff = (1 - cur) * (B_WORDS * 4);
            const int kw = (kk + 1) * 16;
#pragma unroll
            for (int i = 0; i < 4; ++i)
                asm volatile("global_load_async_to_lds_b128 %0, %1, off"
                             :: "v"(lA[i] + aOff), "v"(gA[i] + kw) : "memory");
#pragma unroll
            for (int i = 0; i < 2; ++i)
                asm volatile("global_load_async_to_lds_b128 %0, %1, off"
                             :: "v"(lB[i] + bOff), "v"(gB[i] + kw) : "memory");
            // prefetch two K-steps ahead into L2 (CDNA5 global_prefetch_b8)
            if (kk + 2 < 32) {
                __builtin_prefetch(gA[0] + (kk + 2) * 16, 0, 1);
                __builtin_prefetch(gB[0] + (kk + 2) * 16, 0, 1);
            }
        }

        // --- fragments: two ds_load_b128 each (16B-aligned, conflict-free) --
        // 16-bit operand lane layout (ISA 7.12.2): word for VGPR v =
        //   v<4 -> v + 4*half ; v>=4 -> 8 + (v-4) + 4*half
        const uint4* a4 = reinterpret_cast<const uint4*>(&ldsA[cur][0]);
        const uint4* b4 = reinterpret_cast<const uint4*>(&ldsB[cur][0]);
        FragU af[4], bf[4];
#pragma unroll
        for (int ms = 0; ms < 4; ++ms) {
            int row = wm * 64 + ms * 16 + l15;
            af[ms].q[0] = a4[row * 5 + half];
            af[ms].q[1] = a4[row * 5 + 2 + half];
        }
#pragma unroll
        for (int ns = 0; ns < 4; ++ns) {
            int col = wn * 64 + ns * 16 + l15;
            bf[ns].q[0] = b4[col * 5 + half];
            bf[ns].q[1] = b4[col * 5 + 2 + half];
        }
#pragma unroll
        for (int ms = 0; ms < 4; ++ms)
#pragma unroll
            for (int ns = 0; ns < 4; ++ns)
                acc[ms * 4 + ns] = __builtin_amdgcn_wmma_f32_16x16x32_bf16(
                    false, af[ms].v, false, bf[ns].v, (short)0, acc[ms * 4 + ns],
                    false, false);
    }

    // Store D + bias.  D layout: VGPR r, lanes 0-15 -> M=r, lanes 16-31 -> M=8+r; N = lane&15.
#pragma unroll
    for (int ms = 0; ms < 4; ++ms)
#pragma unroll
        for (int ns = 0; ns < 4; ++ns) {
            int col = N0 + wn * 64 + ns * 16 + l15;
            float bb = bias[col];
#pragma unroll
            for (int r = 0; r < 8; ++r) {
                int mrow = M0 + wm * 64 + ms * 16 + r + 8 * half;
                out[(size_t)mrow * Dm + col] = acc[ms * 4 + ns][r] + bb;
            }
        }
}

// ---------------------------------------------------------------------------
// Dilated attention: one wave32 per (b,h,t); lane owns head dims {2*lane, 2*lane+1}.
// Output written bf16-pair packed to feed the final WMMA GEMM directly.
// ---------------------------------------------------------------------------
__global__ void __launch_bounds__(256)
dmhca_attn(const float* __restrict__ Q, const float* __restrict__ Kf,
           const float* __restrict__ V, unsigned int* __restrict__ Opack) {
    const int lane = threadIdx.x & 31;
    const int wid  = blockIdx.x * 8 + (threadIdx.x >> 5);   // 0 .. 131071
    const int t = wid & (Nseq - 1);
    const int h = (wid >> 11) & (Hh - 1);
    const int b = wid >> 15;

    const size_t rowq = ((size_t)(b * Nseq + t)) * Dm + h * DH;
    const int d0 = lane * 2;
    const float q0 = Q[rowq + d0];
    const float q1 = Q[rowq + d0 + 1];
    const float scale = 0.125f;   // 1/sqrt(64)

    float sc[17];
#pragma unroll
    for (int j = 0; j < 17; ++j) {
        int tk = t + (j - 8) * 2;
        bool ok = (tk >= 0) && (tk < Nseq);
        float p = 0.0f;
        if (ok) {
            size_t kb = ((size_t)(b * Nseq + tk)) * Dm + h * DH + d0;
            p = q0 * Kf[kb] + q1 * Kf[kb + 1];
        }
#pragma unroll
        for (int s = 16; s > 0; s >>= 1) p += __shfl_xor(p, s, 32);
        sc[j] = ok ? p * scale : -INFINITY;
    }

    float mx = sc[0];
#pragma unroll
    for (int j = 1; j < 17; ++j) mx = fmaxf(mx, sc[j]);

    float sum = 0.0f, o0 = 0.0f, o1 = 0.0f;
#pragma unroll
    for (int j = 0; j < 17; ++j) {
        int tk = t + (j - 8) * 2;
        float p = __expf(sc[j] - mx);   // exp(-inf) == 0 for masked slots
        sum += p;
        if ((tk >= 0) && (tk < Nseq)) {
            size_t vb = ((size_t)(b * Nseq + tk)) * Dm + h * DH + d0;
            o0 += p * V[vb];
            o1 += p * V[vb + 1];
        }
    }
    float inv = 1.0f / sum;
    Opack[((size_t)(b * Nseq + t)) * KW + h * 32 + lane] = pack_bf16_rne(o0 * inv, o1 * inv);
}

// ---------------------------------------------------------------------------
extern "C" void kernel_launch(void* const* d_in, const int* in_sizes, int n_in,
                              void* d_out, int out_size, void* d_ws, size_t ws_size,
                              hipStream_t stream) {
    (void)in_sizes; (void)n_in; (void)out_size; (void)ws_size;
    const float* x  = (const float*)d_in[0];
    const float* Wq = (const float*)d_in[1];
    const float* bq = (const float*)d_in[2];
    const float* Wk = (const float*)d_in[3];
    const float* bk = (const float*)d_in[4];
    const float* Wv = (const float*)d_in[5];
    const float* bv = (const float*)d_in[6];
    const float* Wo = (const float*)d_in[7];
    const float* bo = (const float*)d_in[8];
    float* out = (float*)d_out;

    // workspace layout
    char* ws = (char*)d_ws;
    unsigned int* xp    = (unsigned int*)ws;
    size_t off = (size_t)Mtot * KW * 4;
    unsigned int* wpack = (unsigned int*)(ws + off);
    off += (size_t)4 * Dm * KW * 4;
    float* Qf = (float*)(ws + off);  off += (size_t)Mtot * Dm * 4;
    float* Kf = (float*)(ws + off);  off += (size_t)Mtot * Dm * 4;
    float* Vf = (float*)(ws + off);  off += (size_t)Mtot * Dm * 4;
    unsigned int* apack = (unsigned int*)(ws + off);

    unsigned int* wq_p = wpack + (size_t)0 * Dm * KW;
    unsigned int* wk_p = wpack + (size_t)1 * Dm * KW;
    unsigned int* wv_p = wpack + (size_t)2 * Dm * KW;
    unsigned int* wo_p = wpack + (size_t)3 * Dm * KW;

    dmhca_pack_x<<<(Mtot * KW) / 256, 256, 0, stream>>>(x, xp);
    dmhca_pack_w<<<dim3((Dm * KW) / 256, 1, 4), 256, 0, stream>>>(Wq, Wk, Wv, Wo, wpack);

    dim3 gg(Dm / TN, Mtot / TM);
    dmhca_gemm_bf16<<<gg, 256, 0, stream>>>(xp, wq_p, bq, Qf);
    dmhca_gemm_bf16<<<gg, 256, 0, stream>>>(xp, wk_p, bk, Kf);
    dmhca_gemm_bf16<<<gg, 256, 0, stream>>>(xp, wv_p, bv, Vf);

    dmhca_attn<<<(Bq * Hh * Nseq) / 8, 256, 0, stream>>>(Qf, Kf, Vf, apack);

    dmhca_gemm_bf16<<<gg, 256, 0, stream>>>(apack, wo_p, bo, out);
}
